// MSARow_20194936225880
// MI455X (gfx1250) — compile-verified, matrix operand order checked by reading, not run on hardware
//
#include <hip/hip_runtime.h>

// ---------------------------------------------------------------------------
// MSA row attention with pair bias, CDNA5 (gfx1250) BF16 WMMA implementation.
// N=128, L=384, M=256, Z=128, H=8, C=32.
// ---------------------------------------------------------------------------

typedef __bf16 bf16;
typedef __attribute__((ext_vector_type(16))) __bf16 bf16x16;
typedef __attribute__((ext_vector_type(8)))  __bf16 bf16x8;
typedef __attribute__((ext_vector_type(8)))  float  f32x8;
typedef __attribute__((ext_vector_type(4)))  int    i32x4;

#define N_SEQ   128
#define L_RES   384
#define M_DIM   256
#define Z_DIM   128
#define H_HEAD  8
#define C_HEAD  32
#define HC      256          // H*C
#define NC      4096         // N_SEQ*C_HEAD (per-head reduction width)
#define NROWS   49152        // N_SEQ*L_RES
#define LL      147456       // L_RES*L_RES
#define LDS_LD  40           // padded LDS row stride (multiple of 8 -> 16B align)

#define AS_GLOBAL __attribute__((address_space(1)))
#define AS_SHARED __attribute__((address_space(3)))

#if __has_builtin(__builtin_amdgcn_global_load_async_to_lds_b128)
#define HAVE_ASYNC_COPY 1
#else
#define HAVE_ASYNC_COPY 0
#endif
#if __has_builtin(__builtin_amdgcn_s_wait_asynccnt)
#define HAVE_WAIT_ASYNC 1
#else
#define HAVE_WAIT_ASYNC 0
#endif

// 16B global->LDS copy: async (ASYNCcnt-tracked, bypasses VGPRs) if available.
__device__ __forceinline__ void cp16(bf16* lds_dst, const bf16* glb_src) {
#if HAVE_ASYNC_COPY
    __builtin_amdgcn_global_load_async_to_lds_b128(
        (AS_GLOBAL i32x4*)(glb_src),
        (AS_SHARED i32x4*)(lds_dst),
        0, 0);
#else
    *(bf16x8*)lds_dst = *(const bf16x8*)glb_src;
#endif
}

__device__ __forceinline__ void wait_async_all() {
#if HAVE_ASYNC_COPY
#if HAVE_WAIT_ASYNC
    __builtin_amdgcn_s_wait_asynccnt(0);
#else
    asm volatile("s_wait_asynccnt 0" ::: "memory");
#endif
#endif
}

// -------------------------- WMMA fragment helpers --------------------------
// 16-bit A (16x32) / B^T (16 cols x 32 k) fragment: lane&15 = row(col),
// koff = (lane>>4)*8; elements [0..7] = K[koff..koff+7], [8..15] = K[koff+16..+23].
__device__ __forceinline__ bf16x16 ld_frag16(const bf16* base, int lane) {
    const int row  = lane & 15;
    const int koff = (lane >> 4) << 3;
    const bf16* p = base + row * LDS_LD + koff;
    union { bf16x16 f; bf16x8 h[2]; } u;
    u.h[0] = *(const bf16x8*)(p);
    u.h[1] = *(const bf16x8*)(p + 16);
    return u.f;
}

// --------------- BMxBN tile GEMM, 8 waves, double-buffered LDS --------------
// A row-major [BM][K] (lda), B N-major [BN][K] (ldb, i.e. B^T).
// 8 waves in a 4x2 grid; each wave owns (BM/4)x(BN/2): AF=BM/64 A-frags,
// BF=BN/32 B-frags, AF*BF accumulators -> AF*BF WMMAs per 2*(AF+BF)
// ds_load_b128 per 32-k step.
template<int BM, int BN>
__device__ __forceinline__ void gemm_tile(
    const bf16* __restrict__ Ag, int lda,
    const bf16* __restrict__ Bg, int ldb, int K,
    f32x8 (&acc)[BM / 64][BN / 32])
{
    constexpr int AF = BM / 64, BF = BN / 32;
    __shared__ bf16 As[2][BM * LDS_LD];
    __shared__ bf16 Bs[2][BN * LDS_LD];
    const int t    = threadIdx.x;
    const int lane = t & 31;
    const int wave = t >> 5;
    const int wr   = (wave >> 1) * (BM / 4);
    const int wc   = (wave & 1) * (BN / 2);

    auto stage = [&](int k0, int p) {
        constexpr int ASLOT = BM * 4;          // 16B slots in A tile
        constexpr int TOT   = (BM + BN) * 4;   // total 16B slots
#pragma unroll
        for (int s0 = 0; s0 < TOT; s0 += 256) {
            const int s = s0 + t;
            if (s < ASLOT) {
                const int r = s >> 2, kp = (s & 3) * 8;
                cp16(&As[p][r * LDS_LD + kp], Ag + (size_t)r * lda + k0 + kp);
            } else {
                const int s2 = s - ASLOT;
                const int r = s2 >> 2, kp = (s2 & 3) * 8;
                cp16(&Bs[p][r * LDS_LD + kp], Bg + (size_t)r * ldb + k0 + kp);
            }
        }
    };

    stage(0, 0);
    wait_async_all();
    __syncthreads();

    for (int k0 = 0; k0 < K; k0 += 32) {
        const int p = (k0 >> 5) & 1;
        if (k0 + 32 < K) stage(k0 + 32, p ^ 1);   // fill other buffer
        bf16x16 af[AF], bfr[BF];
#pragma unroll
        for (int i = 0; i < AF; ++i)
            af[i] = ld_frag16(&As[p][(wr + 16 * i) * LDS_LD], lane);
#pragma unroll
        for (int j = 0; j < BF; ++j)
            bfr[j] = ld_frag16(&Bs[p][(wc + 16 * j) * LDS_LD], lane);
#pragma unroll
        for (int i = 0; i < AF; ++i)
#pragma unroll
            for (int j = 0; j < BF; ++j)
                acc[i][j] = __builtin_amdgcn_wmma_f32_16x16x32_bf16(
                    false, af[i], false, bfr[j], (short)0, acc[i][j],
                    false, false);
        wait_async_all();
        __syncthreads();
    }
}

// ----------------------------- prep: weights --------------------------------
__global__ __launch_bounds__(256) void k_prep_weights(
    const float* __restrict__ Wq, const float* __restrict__ Wk,
    const float* __restrict__ Wv, const float* __restrict__ Wg,
    const float* __restrict__ Wo, bf16* __restrict__ BT1, bf16* __restrict__ WoT)
{
    int idx = blockIdx.x * 256 + threadIdx.x;
    if (idx < 1024 * 256) {
        int n = idx >> 8, k = idx & 255;
        int id = n >> 8, o = n & 255;
        const float* W = (id == 0) ? Wq : (id == 1) ? Wk : (id == 2) ? Wv : Wg;
        BT1[idx] = (bf16)W[k * 256 + o];
    } else {
        int j = idx - 1024 * 256;
        int n = j >> 8, k = j & 255;
        WoT[j] = (bf16)Wo[k * 256 + n];
    }
}

// ----------------------------- LayerNorm(m) ---------------------------------
__global__ __launch_bounds__(256) void k_ln_m(
    const float* __restrict__ m, const float* __restrict__ gam,
    const float* __restrict__ bet, bf16* __restrict__ mn)
{
    const int lane = threadIdx.x & 31;
    const int row  = blockIdx.x * 8 + (threadIdx.x >> 5);
    const float* x = m + (size_t)row * M_DIM + lane * 8;
    const float4 p0 = *(const float4*)(x);
    const float4 p1 = *(const float4*)(x + 4);
    float v[8] = {p0.x, p0.y, p0.z, p0.w, p1.x, p1.y, p1.z, p1.w};
    float s = 0.f, ss = 0.f;
#pragma unroll
    for (int j = 0; j < 8; ++j) { s += v[j]; ss += v[j] * v[j]; }
#pragma unroll
    for (int d = 16; d >= 1; d >>= 1) {
        s  += __shfl_xor(s,  d, 32);
        ss += __shfl_xor(ss, d, 32);
    }
    const float mu  = s * (1.f / M_DIM);
    const float var = ss * (1.f / M_DIM) - mu * mu;
    const float inv = rsqrtf(var + 1e-5f);
    union { bf16x8 v8; bf16 e[8]; } ou;
#pragma unroll
    for (int j = 0; j < 8; ++j) {
        int c = lane * 8 + j;
        ou.e[j] = (bf16)((v[j] - mu) * inv * gam[c] + bet[c]);
    }
    *(bf16x8*)(mn + (size_t)row * M_DIM + lane * 8) = ou.v8;
}

// ------------------- LayerNorm(z) @ Wz -> pair bias [H][L][L] ---------------
__global__ __launch_bounds__(256) void k_lnz_bias(
    const float* __restrict__ z, const float* __restrict__ gz,
    const float* __restrict__ bz, const float* __restrict__ Wz,
    float* __restrict__ biasB)
{
    const int lane = threadIdx.x & 31;
    const int row  = blockIdx.x * 8 + (threadIdx.x >> 5);   // 0..LL-1
    const float* x = z + (size_t)row * Z_DIM + lane * 4;
    const float4 p = *(const float4*)(x);
    float v[4] = {p.x, p.y, p.z, p.w};
    float s = 0.f, ss = 0.f;
#pragma unroll
    for (int j = 0; j < 4; ++j) { s += v[j]; ss += v[j] * v[j]; }
#pragma unroll
    for (int d = 16; d >= 1; d >>= 1) {
        s  += __shfl_xor(s,  d, 32);
        ss += __shfl_xor(ss, d, 32);
    }
    const float mu  = s * (1.f / Z_DIM);
    const float var = ss * (1.f / Z_DIM) - mu * mu;
    const float inv = rsqrtf(var + 1e-5f);
    float acc[H_HEAD] = {0.f};
#pragma unroll
    for (int j = 0; j < 4; ++j) {
        int c = lane * 4 + j;
        float zn = (v[j] - mu) * inv * gz[c] + bz[c];
#pragma unroll
        for (int h = 0; h < H_HEAD; ++h) acc[h] += zn * Wz[c * H_HEAD + h];
    }
#pragma unroll
    for (int h = 0; h < H_HEAD; ++h)
#pragma unroll
        for (int d = 16; d >= 1; d >>= 1) acc[h] += __shfl_xor(acc[h], d, 32);
    if (lane < H_HEAD)
        biasB[(size_t)lane * LL + row] = acc[lane];
}

// ------------- fused Q/K/V/Gate projection GEMM ([49152x256]@[256x1024]) ----
__global__ __launch_bounds__(256) void k_gemm_qkvg(
    const bf16* __restrict__ mn, const bf16* __restrict__ BT1,
    const float* __restrict__ bg, bf16* __restrict__ Qb,
    bf16* __restrict__ Kb, bf16* __restrict__ Vt, bf16* __restrict__ Gb)
{
    f32x8 acc[2][4] = {};
    const bf16* A = mn  + (size_t)blockIdx.x * 128 * M_DIM;
    const bf16* B = BT1 + (size_t)blockIdx.y * 128 * M_DIM;
    gemm_tile<128, 128>(A, M_DIM, B, M_DIM, M_DIM, acc);

    const int lane  = threadIdx.x & 31;
    const int wave  = threadIdx.x >> 5;
    const int rbase = (wave >> 1) * 32 + ((lane >> 4) << 3);
    const int cbase = (wave & 1) * 64 + (lane & 15);
#pragma unroll
    for (int i = 0; i < 2; ++i)
#pragma unroll
        for (int j = 0; j < 4; ++j) {
            const int gc = blockIdx.y * 128 + cbase + 16 * j;   // 0..1023
            const int id = gc >> 8, cc = gc & 255;
            const int h = cc >> 5, c = cc & 31;
#pragma unroll
            for (int r = 0; r < 8; ++r) {
                const int gr = blockIdx.x * 128 + rbase + 16 * i + r;
                const int b = gr / L_RES, l = gr - b * L_RES;
                const float val = acc[i][j][r];
                if (id == 0)
                    Qb[((size_t)h * L_RES + l) * NC + b * 32 + c] = (bf16)val;
                else if (id == 1)
                    Kb[((size_t)h * L_RES + l) * NC + b * 32 + c] = (bf16)val;
                else if (id == 2)
                    Vt[((size_t)h * NC + b * 32 + c) * L_RES + l] = (bf16)val;
                else {
                    float sg = 1.f / (1.f + __expf(-(val + bg[cc])));
                    Gb[(size_t)gr * HC + cc] = (bf16)sg;
                }
            }
        }
}

// ------------- logits per head: S = Q K^T * (1/64) + bias -------------------
__global__ __launch_bounds__(256) void k_gemm_logits(
    const bf16* __restrict__ Qb, const bf16* __restrict__ Kb,
    const float* __restrict__ biasB, float* __restrict__ Sb)
{
    f32x8 acc[1][2] = {};
    const int h = blockIdx.z;
    const bf16* A = Qb + (size_t)h * L_RES * NC + (size_t)blockIdx.x * 64 * NC;
    const bf16* B = Kb + (size_t)h * L_RES * NC + (size_t)blockIdx.y * 64 * NC;
    gemm_tile<64, 64>(A, NC, B, NC, NC, acc);

    const int lane  = threadIdx.x & 31;
    const int wave  = threadIdx.x >> 5;
    const int rbase = (wave >> 1) * 16 + ((lane >> 4) << 3);
    const int cbase = (wave & 1) * 32 + (lane & 15);
    const float scale = 0.015625f;   // (1/sqrt(C))/sqrt(N) = 1/64
#pragma unroll
    for (int j = 0; j < 2; ++j) {
        const int gv = blockIdx.y * 64 + cbase + 16 * j;
#pragma unroll
        for (int r = 0; r < 8; ++r) {
            const int gq = blockIdx.x * 64 + rbase + r;
            const size_t o = ((size_t)h * L_RES + gq) * L_RES + gv;
            Sb[o] = acc[0][j][r] * scale + biasB[o];
        }
    }
}

// ----------------------------- softmax over v -------------------------------
__global__ __launch_bounds__(256) void k_softmax(
    const float* __restrict__ Sb, bf16* __restrict__ Pb)
{
    const int lane = threadIdx.x & 31;
    const int row  = blockIdx.x * 8 + (threadIdx.x >> 5);   // 0..H*L-1
    const float* x = Sb + (size_t)row * L_RES;
    float v[12];
    float mx = -3.4e38f;
#pragma unroll
    for (int i = 0; i < 12; ++i) { v[i] = x[lane + 32 * i]; mx = fmaxf(mx, v[i]); }
#pragma unroll
    for (int d = 16; d >= 1; d >>= 1) mx = fmaxf(mx, __shfl_xor(mx, d, 32));
    float s = 0.f;
#pragma unroll
    for (int i = 0; i < 12; ++i) { v[i] = __expf(v[i] - mx); s += v[i]; }
#pragma unroll
    for (int d = 16; d >= 1; d >>= 1) s += __shfl_xor(s, d, 32);
    const float inv = 1.f / s;
#pragma unroll
    for (int i = 0; i < 12; ++i)
        Pb[(size_t)row * L_RES + lane + 32 * i] = (bf16)(v[i] * inv);
}

// ---------------- O = P @ V per head, fused sigmoid gating ------------------
__global__ __launch_bounds__(256) void k_gemm_pv(
    const bf16* __restrict__ Pb, const bf16* __restrict__ Vt,
    const bf16* __restrict__ Gb, bf16* __restrict__ Ofull)
{
    f32x8 acc[2][4] = {};
    const int h = blockIdx.z;
    const bf16* A = Pb + (size_t)h * LL + (size_t)blockIdx.x * 128 * L_RES;
    const bf16* B = Vt + (size_t)h * NC * L_RES + (size_t)blockIdx.y * 128 * L_RES;
    gemm_tile<128, 128>(A, L_RES, B, L_RES, L_RES, acc);

    const int lane  = threadIdx.x & 31;
    const int wave  = threadIdx.x >> 5;
    const int rbase = (wave >> 1) * 32 + ((lane >> 4) << 3);
    const int cbase = (wave & 1) * 64 + (lane & 15);
#pragma unroll
    for (int i = 0; i < 2; ++i)
#pragma unroll
        for (int j = 0; j < 4; ++j) {
            const int gn = blockIdx.y * 128 + cbase + 16 * j;  // 0..4095
            const int b = gn >> 5, c = gn & 31;
#pragma unroll
            for (int r = 0; r < 8; ++r) {
                const int gq = blockIdx.x * 128 + rbase + 16 * i + r;
                const size_t idx = ((size_t)b * L_RES + gq) * HC + h * 32 + c;
                const float gate = (float)Gb[idx];
                Ofull[idx] = (bf16)(acc[i][j][r] * gate);
            }
        }
}

// ------------------- out = (O*g) @ Wo + bo  (f32 result) --------------------
__global__ __launch_bounds__(256) void k_gemm_out(
    const bf16* __restrict__ Ofull, const bf16* __restrict__ WoT,
    const float* __restrict__ bo, float* __restrict__ outp)
{
    f32x8 acc[2][4] = {};
    const bf16* A = Ofull + (size_t)blockIdx.x * 128 * HC;
    const bf16* B = WoT   + (size_t)blockIdx.y * 128 * HC;
    gemm_tile<128, 128>(A, HC, B, HC, HC, acc);

    const int lane  = threadIdx.x & 31;
    const int wave  = threadIdx.x >> 5;
    const int rbase = (wave >> 1) * 32 + ((lane >> 4) << 3);
    const int cbase = (wave & 1) * 64 + (lane & 15);
#pragma unroll
    for (int i = 0; i < 2; ++i)
#pragma unroll
        for (int j = 0; j < 4; ++j) {
            const int gc = blockIdx.y * 128 + cbase + 16 * j;
#pragma unroll
            for (int r = 0; r < 8; ++r) {
                const int gr = blockIdx.x * 128 + rbase + 16 * i + r;
                outp[(size_t)gr * M_DIM + gc] = acc[i][j][r] + bo[gc];
            }
        }
}

// ------------------------------- launcher -----------------------------------
extern "C" void kernel_launch(void* const* d_in, const int* in_sizes, int n_in,
                              void* d_out, int out_size, void* d_ws, size_t ws_size,
                              hipStream_t stream) {
    const float* m     = (const float*)d_in[0];
    const float* z     = (const float*)d_in[1];
    const float* ln1_g = (const float*)d_in[2];
    const float* ln1_b = (const float*)d_in[3];
    const float* Wq    = (const float*)d_in[4];
    const float* Wk    = (const float*)d_in[5];
    const float* Wv    = (const float*)d_in[6];
    const float* lnz_g = (const float*)d_in[7];
    const float* lnz_b = (const float*)d_in[8];
    const float* Wz    = (const float*)d_in[9];
    const float* Wg    = (const float*)d_in[10];
    const float* bg    = (const float*)d_in[11];
    const float* Wo    = (const float*)d_in[12];
    const float* bo    = (const float*)d_in[13];
    float* outp = (float*)d_out;

    char* w = (char*)d_ws;
    size_t off = 0;
    auto alloc = [&](size_t bytes) -> void* {
        void* p = w + off;
        off = (off + bytes + 255) & ~(size_t)255;
        return p;
    };
    bf16*  mn    = (bf16*)alloc((size_t)NROWS * M_DIM * 2);
    bf16*  BT1   = (bf16*)alloc((size_t)1024 * 256 * 2);
    bf16*  WoT   = (bf16*)alloc((size_t)256 * 256 * 2);
    bf16*  Qb    = (bf16*)alloc((size_t)H_HEAD * L_RES * NC * 2);
    bf16*  Kb    = (bf16*)alloc((size_t)H_HEAD * L_RES * NC * 2);
    bf16*  Vt    = (bf16*)alloc((size_t)H_HEAD * NC * L_RES * 2);
    bf16*  Gb    = (bf16*)alloc((size_t)NROWS * HC * 2);
    float* biasB = (float*)alloc((size_t)H_HEAD * LL * 4);
    float* Sb    = (float*)alloc((size_t)H_HEAD * LL * 4);
    bf16*  Pb    = (bf16*)alloc((size_t)H_HEAD * LL * 2);
    bf16*  Ofull = (bf16*)alloc((size_t)NROWS * HC * 2);

    k_prep_weights<<<1280, 256, 0, stream>>>(Wq, Wk, Wv, Wg, Wo, BT1, WoT);
    k_ln_m<<<NROWS / 8, 256, 0, stream>>>(m, ln1_g, ln1_b, mn);
    k_lnz_bias<<<LL / 8, 256, 0, stream>>>(z, lnz_g, lnz_b, Wz, biasB);
    k_gemm_qkvg<<<dim3(NROWS / 128, 1024 / 128), 256, 0, stream>>>(
        mn, BT1, bg, Qb, Kb, Vt, Gb);
    k_gemm_logits<<<dim3(L_RES / 64, L_RES / 64, H_HEAD), 256, 0, stream>>>(
        Qb, Kb, biasB, Sb);
    k_softmax<<<(H_HEAD * L_RES) / 8, 256, 0, stream>>>(Sb, Pb);
    k_gemm_pv<<<dim3(L_RES / 128, NC / 128, H_HEAD), 256, 0, stream>>>(
        Pb, Vt, Gb, Ofull);
    k_gemm_out<<<dim3(NROWS / 128, M_DIM / 128), 256, 0, stream>>>(
        Ofull, WoT, bo, outp);
}
